// Stage_35579509080778
// MI455X (gfx1250) — compile-verified, matrix operand order checked by reading, not run on hardware
//
#include <hip/hip_runtime.h>
#include <hip/hip_bf16.h>

// ---------------------------------------------------------------------------
// Point-cloud block forward for MI455X (gfx1250): wave32, WMMA bf16, TDM
// weight staging into LDS. N=32768, K=24, DIM=128, HEAD=256, 8 segments.
// ---------------------------------------------------------------------------

#define N_PTS 32768
#define KNBR  24
#define DIMC  128
#define HEADC 256
#define NSEG  8
#define EPSV  1e-5f

typedef unsigned short u16;
typedef __attribute__((ext_vector_type(16))) __bf16 v16bf;
typedef __attribute__((ext_vector_type(8)))  float  v8f;
typedef __attribute__((ext_vector_type(4)))  unsigned int u32x4;
typedef __attribute__((ext_vector_type(8)))  int  i32x8;
typedef __attribute__((ext_vector_type(4)))  int  i32x4;

#define EP_NONE      0
#define EP_BIAS_GELU 1
#define EP_SS_ADD    2

#if defined(__has_builtin)
#  if __has_builtin(__builtin_amdgcn_tensor_load_to_lds)
#    define HAVE_TDM 1
#  endif
#endif
#ifndef HAVE_TDM
#  define HAVE_TDM 0
#endif

// ------------------------------- device helpers ----------------------------

__device__ __forceinline__ float gelu_tanh(float x) {
  float x3 = x * x * x;
  return 0.5f * x * (1.0f + tanhf(0.79788456080286535588f * (x + 0.044715f * x3)));
}

__device__ __forceinline__ u16 bf16bits(float f) {
  union { __bf16 h; u16 u; } c;
  c.h = (__bf16)f;
  return c.u;
}

__device__ __forceinline__ v16bf cvt16(const float* t) {
  v16bf v;
#pragma unroll
  for (int i = 0; i < 16; ++i) v[i] = (__bf16)t[i];
  return v;
}

// A fragment (16x32 bf16 tile) from row-major f32: lane<16 -> row=lane, K in
// {k0..k0+7, k0+16..k0+23}; lane>=16 -> row=lane-16, K base +8 (ISA 7.12.2).
__device__ __forceinline__ v16bf load_a_f32(const float* __restrict__ A, int lda,
                                            int row0, int k0, int lane) {
  int half = lane >> 4;
  const float* p = A + (long)(row0 + (lane & 15)) * lda + k0 + half * 8;
  float4 a0 = *(const float4*)(p);
  float4 a1 = *(const float4*)(p + 4);
  float4 b0 = *(const float4*)(p + 16);
  float4 b1 = *(const float4*)(p + 20);
  float t[16] = {a0.x,a0.y,a0.z,a0.w, a1.x,a1.y,a1.z,a1.w,
                 b0.x,b0.y,b0.z,b0.w, b1.x,b1.y,b1.z,b1.w};
  return cvt16(t);
}

// B fragment from bf16 weights Bt[out][in] (= B^T row-major), global or LDS:
// lane gives column N, same K grouping as A -> two b128 loads per lane.
__device__ __forceinline__ v16bf load_b_bf16(const u16* __restrict__ Bt, int ldb,
                                             int col0, int k0, int lane) {
  int half = lane >> 4;
  const u16* p = Bt + (long)(col0 + (lane & 15)) * ldb + k0 + half * 8;
  union { uint4 q[2]; v16bf v; } u;
  u.q[0] = *(const uint4*)(p);
  u.q[1] = *(const uint4*)(p + 16);
  return u.v;
}

__device__ __forceinline__ v8f wmma_bf16(v16bf a, v16bf b, v8f c) {
  return __builtin_amdgcn_wmma_f32_16x16x32_bf16(false, a, false, b, (short)0, c,
                                                 false, false);
}

#if HAVE_TDM
// Issue a 2D Tensor-DMA load of a [rows x k_elems] bf16 matrix (row-major,
// stride = k_elems) from global memory into LDS at lds_off. D# per ISA §8:
// group0 = {count=1, lds_addr, global_addr, type=2}, group1 packs data_size=2B,
// tensor/tile dims and dim0 stride; groups 2/3 zero (2D tensor).
// This toolchain's builtin takes 6 args (g0, g1, g2, g3, g4, cpol).
__device__ __forceinline__ void tdm_load_2d_bf16(const void* gaddr, unsigned lds_off,
                                                 unsigned k_elems, unsigned rows) {
  unsigned long long ga = (unsigned long long)gaddr;
  u32x4 g0;
  g0[0] = 1u;                                                  // count=1, user mode
  g0[1] = lds_off;                                             // lds_addr (bytes)
  g0[2] = (unsigned)(ga & 0xFFFFFFFFu);                        // global_addr[31:0]
  g0[3] = (unsigned)((ga >> 32) & 0x1FFFFFFu) | (2u << 30);    // addr[56:32] | type=2
  i32x8 g1;
  g1[0] = (int)(1u << 16);                                     // data_size=1 (2 bytes)
  g1[1] = (int)((k_elems & 0xFFFFu) << 16);                    // tensor_dim0 lo
  g1[2] = (int)(((k_elems >> 16) & 0xFFFFu) | ((rows & 0xFFFFu) << 16)); // d0 hi|d1 lo
  g1[3] = (int)(((rows >> 16) & 0xFFFFu) | ((k_elems & 0xFFFFu) << 16)); // d1 hi|tile_d0
  g1[4] = (int)(rows & 0xFFFFu);                               // tile_dim1 (tile_dim2=0)
  g1[5] = (int)k_elems;                                        // tensor_dim0_stride lo
  g1[6] = 0;
  g1[7] = 0;
  i32x4 z4 = {0, 0, 0, 0};
  i32x8 z8 = {0, 0, 0, 0, 0, 0, 0, 0};
  __builtin_amdgcn_tensor_load_to_lds(g0, g1, z4, z4, z8, 0);
}
#endif

// ------------------------------- prep kernels ------------------------------

__global__ void k_transpose_bf16(const float* __restrict__ src, u16* __restrict__ dst,
                                 int I, int O) {
  long idx = (long)blockIdx.x * blockDim.x + threadIdx.x;
  if (idx >= (long)I * O) return;
  int i = (int)(idx / O), o = (int)(idx % O);
  dst[(long)o * I + i] = bf16bits(src[(long)i * O + o]);
}

__global__ void k_bn_fuse(const float* __restrict__ g, const float* __restrict__ b,
                          const float* __restrict__ m, const float* __restrict__ v,
                          float* __restrict__ ss, int C) {
  int c = blockIdx.x * blockDim.x + threadIdx.x;
  if (c >= C) return;
  float s = g[c] * rsqrtf(v[c] + EPSV);
  ss[c]     = s;
  ss[C + c] = b[c] - m[c] * s;
}

__global__ void k_scale_shift(const float* __restrict__ in, const float* __restrict__ ss,
                              float* __restrict__ out, int C, long total) {
  long i = (long)blockIdx.x * blockDim.x + threadIdx.x;
  if (i >= total) return;
  int c = (int)(i % C);
  out[i] = in[i] * ss[c] + ss[C + c];
}

// ------------------------------- x_embed (VALU, tiny) -----------------------

__global__ void k_x_embed(const float* __restrict__ x,
                          const float* __restrict__ w1, const float* __restrict__ ss1,
                          const float* __restrict__ w2, const float* __restrict__ ss2,
                          const float* __restrict__ w3, float* __restrict__ xem) {
  __shared__ float h1[16];
  __shared__ float h2[32];
  int p = blockIdx.x;
  int t = threadIdx.x;
  float4 xp = *(const float4*)(x + (long)p * 4);
  if (t < 16) {
    float v = xp.x * w1[0*16+t] + xp.y * w1[1*16+t] + xp.z * w1[2*16+t] + xp.w * w1[3*16+t];
    h1[t] = gelu_tanh(v * ss1[t] + ss1[16 + t]);
  }
  __syncthreads();
  if (t < 32) {
    float v = 0.f;
#pragma unroll
    for (int j = 0; j < 16; ++j) v += h1[j] * w2[j*32 + t];
    h2[t] = gelu_tanh(v * ss2[t] + ss2[32 + t]);
  }
  __syncthreads();
  float v = 0.f;
#pragma unroll
  for (int j = 0; j < 32; ++j) v += h2[j] * w3[j*DIMC + t];
  xem[(long)p * DIMC + t] = v;
}

// ---------------- fused neighbor embed + gather + max (WMMA) ----------------

__global__ void __launch_bounds__(256)
k_nbr_fused(const float* __restrict__ x, const float* __restrict__ xyz,
            const int* __restrict__ knn, const float* __restrict__ xem,
            const u16* __restrict__ w3t,
            const float* __restrict__ w1, const float* __restrict__ ss1,
            const float* __restrict__ w2, const float* __restrict__ ss2,
            const float* __restrict__ nbrss, float* __restrict__ h0) {
  __shared__ __align__(16) u16 aShared[8][32 * 32];
  int lane = threadIdx.x & 31;
  int wave = threadIdx.x >> 5;
  int wpb  = blockDim.x >> 5;
  int half = lane >> 4, l15 = lane & 15;

  v16bf bfrag[8];
#pragma unroll
  for (int n = 0; n < 8; ++n) bfrag[n] = load_b_bf16(w3t, 32, n * 16, 0, lane);

  for (long p = (long)blockIdx.x * wpb + wave; p < N_PTS; p += (long)gridDim.x * wpb) {
    float xi0 = xyz[p*3+0], xi1 = xyz[p*3+1], xi2 = xyz[p*3+2];
    u16* aw = &aShared[wave][0];

    for (int k = 0; k < KNBR; ++k) {
      int j = knn[p * KNBR + k];
      float f0 = xyz[j*3+0] - xi0, f1 = xyz[j*3+1] - xi1, f2 = xyz[j*3+2] - xi2;
      float4 xj = *(const float4*)(x + (long)j * 4);
      int c1 = l15;
      float h1 = f0*w1[0*16+c1] + f1*w1[1*16+c1] + f2*w1[2*16+c1]
               + xj.x*w1[3*16+c1] + xj.y*w1[4*16+c1] + xj.z*w1[5*16+c1] + xj.w*w1[6*16+c1];
      h1 = gelu_tanh(h1 * ss1[c1] + ss1[16 + c1]);
      float h2 = 0.f;
#pragma unroll
      for (int jj = 0; jj < 16; ++jj) {
        float hj = __shfl(h1, jj, 32);
        h2 += hj * w2[jj*32 + lane];
      }
      h2 = gelu_tanh(h2 * ss2[lane] + ss2[32 + lane]);
      aw[k * 32 + lane] = bf16bits(h2);
    }
#pragma unroll
    for (int k = KNBR; k < 32; ++k) aw[k * 32 + lane] = 0;
    asm volatile("s_wait_dscnt 0x0" ::: "memory");

    float colmax[8];
#pragma unroll
    for (int n = 0; n < 8; ++n) colmax[n] = -3.0e38f;

    for (int rtile = 0; rtile < 2; ++rtile) {
      const u16* ap = aw + (rtile * 16 + l15) * 32 + half * 8;
      union { uint4 q[2]; v16bf v; } ua;
      ua.q[0] = *(const uint4*)(ap);
      ua.q[1] = *(const uint4*)(ap + 16);
#pragma unroll
      for (int n = 0; n < 8; ++n) {
        v8f c;
#pragma unroll
        for (int v = 0; v < 8; ++v) {
          int kr = rtile * 16 + v + half * 8;
          if (kr < KNBR) {
            int j = knn[p * KNBR + kr];
            c[v] = xem[(long)j * DIMC + n * 16 + l15];
          } else {
            c[v] = -3.0e38f;
          }
        }
        v8f d = wmma_bf16(ua.v, bfrag[n], c);
        float m = d[0];
#pragma unroll
        for (int v = 1; v < 8; ++v) m = fmaxf(m, d[v]);
        m = fmaxf(m, __shfl_xor(m, 16, 32));
        colmax[n] = fmaxf(colmax[n], m);
      }
    }
    if (half == 0) {
#pragma unroll
      for (int n = 0; n < 8; ++n) {
        int c = n * 16 + l15;
        h0[p * DIMC + c] = colmax[n] * nbrss[c] + nbrss[DIMC + c];
      }
    }
  }
}

// ------------------- WMMA GEMM with TDM-staged weights ----------------------
// Out[M,NC] = epilogue(A[M,K] @ Bt[NC,K]^T).  Block = 8 waves = 128 rows.
// Weights (NC*K bf16, <=64KB) are DMA'd into LDS once per block by the Tensor
// Data Mover; each wave preloads its 16-row A strip into VGPRs and sweeps all
// column tiles from LDS.

template <int K, int NC, int EP>
__global__ void __launch_bounds__(256)
k_gemm16t(const float* __restrict__ A, const u16* __restrict__ Bt,
          float* __restrict__ Out, const float* __restrict__ Res,
          const float* __restrict__ p0, const float* __restrict__ p1, int M) {
  __shared__ __align__(16) u16 bS[NC * K];
  int lane = threadIdx.x & 31;
  int wave = threadIdx.x >> 5;

#if HAVE_TDM
  if (wave == 0) {
    tdm_load_2d_bf16(Bt, (unsigned)(unsigned long long)&bS[0], (unsigned)K, (unsigned)NC);
    __builtin_amdgcn_s_wait_tensorcnt(0);
  }
#else
  for (int i = threadIdx.x; i < (NC * K) / 8; i += blockDim.x)
    ((uint4*)bS)[i] = ((const uint4*)Bt)[i];
#endif
  __syncthreads();

  int row0 = blockIdx.x * 128 + wave * 16;   // grid sized so row0 < M always

  // Preload A fragments for this 16-row strip (reused across all NC tiles).
  v16bf afrag[K / 32];
#pragma unroll
  for (int kk = 0; kk < K / 32; ++kk) afrag[kk] = load_a_f32(A, K, row0, kk * 32, lane);

  int col15 = lane & 15;
  int rbase = row0 + ((lane >> 4) ? 8 : 0);
#pragma unroll
  for (int ct = 0; ct < NC / 16; ++ct) {
    v8f acc = {0.f, 0.f, 0.f, 0.f, 0.f, 0.f, 0.f, 0.f};
#pragma unroll
    for (int kk = 0; kk < K / 32; ++kk) {
      v16bf b = load_b_bf16(&bS[0], K, ct * 16, kk * 32, lane);
      acc = wmma_bf16(afrag[kk], b, acc);
    }
    int col = ct * 16 + col15;
#pragma unroll
    for (int v = 0; v < 8; ++v) {
      int r = rbase + v;
      float d = acc[v];
      float y;
      if (EP == EP_BIAS_GELU)      y = gelu_tanh(d + p0[col]);
      else if (EP == EP_SS_ADD)    y = d * p0[col] + p1[col] + Res[(long)r * NC + col];
      else                         y = d;
      Out[(long)r * NC + col] = y;
    }
  }
  (void)M;
}

// --------------------- LFP gather-max / subtract / bn / add -----------------

__global__ void __launch_bounds__(256)
k_lfp_gather(const float* __restrict__ t1, const int* __restrict__ knn,
             const float* __restrict__ ss, float* __restrict__ h) {
  int lane = threadIdx.x & 31;
  int wave = threadIdx.x >> 5;
  int wpb  = blockDim.x >> 5;
  for (long p = (long)blockIdx.x * wpb + wave; p < N_PTS; p += (long)gridDim.x * wpb) {
    int c0 = lane * 4;
    float4 ti = *(const float4*)(t1 + p * DIMC + c0);
    float4 m  = make_float4(-3.0e38f, -3.0e38f, -3.0e38f, -3.0e38f);
    for (int k = 0; k < KNBR; ++k) {
      int j = knn[p * KNBR + k];
      float4 tj = *(const float4*)(t1 + (long)j * DIMC + c0);
      m.x = fmaxf(m.x, tj.x); m.y = fmaxf(m.y, tj.y);
      m.z = fmaxf(m.z, tj.z); m.w = fmaxf(m.w, tj.w);
    }
    float4 hi = *(const float4*)(h + p * DIMC + c0);
    float4 o;
    o.x = hi.x + (m.x - ti.x) * ss[c0+0] + ss[DIMC + c0+0];
    o.y = hi.y + (m.y - ti.y) * ss[c0+1] + ss[DIMC + c0+1];
    o.z = hi.z + (m.z - ti.z) * ss[c0+2] + ss[DIMC + c0+2];
    o.w = hi.w + (m.w - ti.w) * ss[c0+3] + ss[DIMC + c0+3];
    *(float4*)(h + p * DIMC + c0) = o;
  }
}

// ------------------------------- dcd head -----------------------------------

__global__ void k_seg_partial(const float* __restrict__ g, float* __restrict__ part) {
  int s = blockIdx.x, chunk = blockIdx.y, c = threadIdx.x;
  const int rows = (N_PTS / NSEG) / 16;   // 256 rows per chunk
  const float* base = g + ((long)s * (N_PTS / NSEG) + (long)chunk * rows) * HEADC + c;
  float acc = 0.f;
  for (int r = 0; r < rows; ++r) acc += base[(long)r * HEADC];
  part[(s * 16 + chunk) * HEADC + c] = acc;
}

__global__ void k_seg_final(const float* __restrict__ part, float* __restrict__ means) {
  int s = blockIdx.x, c = threadIdx.x;
  float acc = 0.f;
#pragma unroll
  for (int k = 0; k < 16; ++k) acc += part[(s * 16 + k) * HEADC + c];
  means[s * HEADC + c] = acc * (1.0f / (N_PTS / NSEG));
}

__global__ void k_dcd_fc(const float* __restrict__ means, const float* __restrict__ W1,
                         const float* __restrict__ bnss, const float* __restrict__ W2,
                         float* __restrict__ outv, int do_sigmoid) {
  __shared__ float hid[NSEG][1024];
  int t = threadIdx.x;                    // blockDim = 1024
  for (int s = 0; s < NSEG; ++s) {
    float v = 0.f;
    for (int j = 0; j < HEADC; ++j) v += means[s * HEADC + j] * W1[j * 1024 + t];
    hid[s][t] = gelu_tanh(v * bnss[t] + bnss[1024 + t]);
  }
  __syncthreads();
  if (t < HEADC) {
    for (int s = 0; s < NSEG; ++s) {
      float v = 0.f;
      for (int j = 0; j < 1024; ++j) v += hid[s][j] * W2[j * HEADC + t];
      v = gelu_tanh(v);
      if (do_sigmoid) v = 1.0f / (1.0f + expf(-v));
      outv[s * HEADC + t] = v;
    }
  }
}

__global__ void k_dcd_final(const float* __restrict__ g, const int* __restrict__ seg,
                            const float* __restrict__ outm, const float* __restrict__ outw,
                            float* __restrict__ out) {
  long i = (long)blockIdx.x * blockDim.x + threadIdx.x;
  if (i >= (long)N_PTS * HEADC) return;
  int r = (int)(i / HEADC), c = (int)(i % HEADC);
  int s = seg[r];
  float xv = g[i];
  out[i] = outw[s * HEADC + c] * xv * 0.5f + 0.75f * xv + outm[s * HEADC + c];
}

// ------------------------------- host launcher ------------------------------

extern "C" void kernel_launch(void* const* d_in, const int* in_sizes, int n_in,
                              void* d_out, int out_size, void* d_ws, size_t ws_size,
                              hipStream_t stream) {
  (void)in_sizes; (void)n_in; (void)out_size; (void)ws_size;
  // input order: x, xyz, knn, seg_ids, then jax-pytree (alphabetical) params
  const float* X    = (const float*)d_in[0];
  const float* XYZ  = (const float*)d_in[1];
  const int*   KNNP = (const int*)d_in[2];
  const int*   SEG  = (const int*)d_in[3];
  int q = 4;
  auto F = [&]() { return (const float*)d_in[q++]; };

  const float *lfpW[4], *lfpBN[4][4];
  for (int j = 0; j < 4; ++j) { lfpW[j] = F(); for (int t = 0; t < 4; ++t) lfpBN[j][t] = F(); }
  const float *mW1[3], *mW2[3], *mb1[3], *mBN[3][4];   // [0]=mlp0, [1..2]=mlps
  for (int j = 0; j < 3; ++j) { mW1[j] = F(); mW2[j] = F(); mb1[j] = F();
                                for (int t = 0; t < 4; ++t) mBN[j][t] = F(); }
  const float *fcW1[2], *fcW2[2], *fcBN[2][4];
  for (int j = 0; j < 2; ++j) { fcW1[j] = F(); fcW2[j] = F();
                                for (int t = 0; t < 4; ++t) fcBN[j][t] = F(); }
  const float* nbrBN[4]; for (int t = 0; t < 4; ++t) nbrBN[t] = F();
  const float *neW1 = F(), *neW2 = F(), *neW3 = F();
  const float *neBN1[4]; for (int t = 0; t < 4; ++t) neBN1[t] = F();
  const float *neBN2[4]; for (int t = 0; t < 4; ++t) neBN2[t] = F();
  const float *ppW = F(); const float* ppBN[4]; for (int t = 0; t < 4; ++t) ppBN[t] = F();
  const float *xeW1 = F(), *xeW2 = F(), *xeW3 = F();
  const float *xeBN1[4]; for (int t = 0; t < 4; ++t) xeBN1[t] = F();
  const float *xeBN2[4]; for (int t = 0; t < 4; ++t) xeBN2[t] = F();

  // ---- workspace layout
  char* W = (char*)d_ws;
  size_t off = 0;
  auto alloc = [&](size_t bytes) -> void* {
    off = (off + 255) & ~(size_t)255;
    void* p = W + off; off += bytes; return p;
  };
  float* XEM  = (float*)alloc((size_t)N_PTS * DIMC * 4);
  float* H    = (float*)alloc((size_t)N_PTS * DIMC * 4);
  float* T1   = (float*)alloc((size_t)N_PTS * DIMC * 4);
  float* HID  = (float*)alloc((size_t)N_PTS * HEADC * 4);
  float* G    = (float*)alloc((size_t)N_PTS * HEADC * 4);
  float* PART = (float*)alloc(NSEG * 16 * HEADC * 4);
  float* MEANS= (float*)alloc(NSEG * HEADC * 4);
  float* OUTM = (float*)alloc(NSEG * HEADC * 4);
  float* OUTW = (float*)alloc(NSEG * HEADC * 4);
  u16* W3T    = (u16*)alloc(128 * 32 * 2);
  u16* LFPWT[4]; for (int j = 0; j < 4; ++j) LFPWT[j] = (u16*)alloc(128 * 128 * 2);
  u16* MW1T[3];  for (int j = 0; j < 3; ++j) MW1T[j]  = (u16*)alloc(256 * 128 * 2);
  u16* MW2T[3];  for (int j = 0; j < 3; ++j) MW2T[j]  = (u16*)alloc(128 * 256 * 2);
  u16* POSTWT = (u16*)alloc(256 * 128 * 2);
  float* XE1SS = (float*)alloc(32 * 4);
  float* XE2SS = (float*)alloc(64 * 4);
  float* NE1SS = (float*)alloc(32 * 4);
  float* NE2SS = (float*)alloc(64 * 4);
  float* NBRSS = (float*)alloc(256 * 4);
  float* MLPSS[3]; for (int j = 0; j < 3; ++j) MLPSS[j] = (float*)alloc(256 * 4);
  float* LFPSS[4]; for (int j = 0; j < 4; ++j) LFPSS[j] = (float*)alloc(256 * 4);
  float* POSTSS = (float*)alloc(256 * 4);
  float* FCSS  = (float*)alloc(2048 * 4);
  float* FC1SS = (float*)alloc(2048 * 4);

  // ---- prep: fold BN into scale/shift, transpose weights to bf16 [out][in]
  auto bnf = [&](const float* const bn[4], float* ss, int C) {
    k_bn_fuse<<<(C + 255) / 256, 256, 0, stream>>>(bn[1], bn[0], bn[2], bn[3], ss, C);
  };
  bnf(xeBN1, XE1SS, 16);  bnf(xeBN2, XE2SS, 32);
  bnf(neBN1, NE1SS, 16);  bnf(neBN2, NE2SS, 32);
  bnf(nbrBN, NBRSS, 128);
  for (int j = 0; j < 3; ++j) bnf(mBN[j], MLPSS[j], 128);
  for (int j = 0; j < 4; ++j) bnf(lfpBN[j], LFPSS[j], 128);
  bnf(ppBN, POSTSS, 128);
  bnf(fcBN[0], FCSS, 1024);  bnf(fcBN[1], FC1SS, 1024);

  auto tr = [&](const float* src, u16* dst, int I, int O) {
    long n = (long)I * O;
    k_transpose_bf16<<<(int)((n + 255) / 256), 256, 0, stream>>>(src, dst, I, O);
  };
  tr(neW3, W3T, 32, 128);
  for (int j = 0; j < 4; ++j) tr(lfpW[j], LFPWT[j], 128, 128);
  for (int j = 0; j < 3; ++j) { tr(mW1[j], MW1T[j], 128, 256); tr(mW2[j], MW2T[j], 256, 128); }
  tr(ppW, POSTWT, 128, 256);

  // ---- x_embed, fused neighbor stage -> H = h0[N,128]
  k_x_embed<<<N_PTS, 128, 0, stream>>>(X, xeW1, XE1SS, xeW2, XE2SS, xeW3, XEM);
  k_nbr_fused<<<512, 256, 0, stream>>>(X, XYZ, KNNP, XEM, W3T,
                                       neW1, NE1SS, neW2, NE2SS, NBRSS, H);

  const int GB = N_PTS / 128;   // 256 blocks, 128 rows each
  auto mlp = [&](int j) {
    k_gemm16t<128, 256, EP_BIAS_GELU><<<GB, 256, 0, stream>>>(
        H, MW1T[j], HID, nullptr, mb1[j], nullptr, N_PTS);
    k_gemm16t<256, 128, EP_SS_ADD><<<GB, 256, 0, stream>>>(
        HID, MW2T[j], H, H, MLPSS[j], MLPSS[j] + 128, N_PTS);
  };
  auto lfp = [&](int j) {
    k_gemm16t<128, 128, EP_NONE><<<GB, 256, 0, stream>>>(
        H, LFPWT[j], T1, nullptr, nullptr, nullptr, N_PTS);
    k_lfp_gather<<<1024, 256, 0, stream>>>(T1, KNNP, LFPSS[j], H);
  };

  // block: x += mlp0; [lfp0; lfp1; mlps0; lfp2; lfp3; mlps1]
  mlp(0);
  lfp(0);
  lfp(1); mlp(1);
  lfp(2);
  lfp(3); mlp(2);

  // postproj: bn(H) @ W -> G[N,256]
  k_scale_shift<<<(int)(((long)N_PTS * DIMC + 255) / 256), 256, 0, stream>>>(
      H, POSTSS, T1, DIMC, (long)N_PTS * DIMC);
  k_gemm16t<128, 256, EP_NONE><<<GB, 256, 0, stream>>>(
      T1, POSTWT, G, nullptr, nullptr, nullptr, N_PTS);

  // dcd
  k_seg_partial<<<dim3(NSEG, 16), HEADC, 0, stream>>>(G, PART);
  k_seg_final<<<NSEG, HEADC, 0, stream>>>(PART, MEANS);
  k_dcd_fc<<<1, 1024, 0, stream>>>(MEANS, fcW1[0], FCSS, fcW2[0], OUTM, 0);
  k_dcd_fc<<<1, 1024, 0, stream>>>(MEANS, fcW1[1], FC1SS, fcW2[1], OUTW, 1);
  k_dcd_final<<<(int)(((long)N_PTS * HEADC + 255) / 256), 256, 0, stream>>>(
      G, SEG, OUTM, OUTW, (float*)d_out);
}
